// MHA_48713519071709
// MI455X (gfx1250) — matrix operand order repaired and checked by reading, not compiled, not run on hardware
//
#include <hip/hip_runtime.h>
#include <stdint.h>

// ---------------- types for WMMA ----------------
typedef __bf16 bf16_t;
typedef bf16_t v16bf __attribute__((ext_vector_type(16)));
typedef float  v8f   __attribute__((ext_vector_type(8)));

union FragBF {                 // 16 bf16 elements = 2 x 16B chunks
    uint4    q[2];
    uint32_t u[8];
    v16bf    v;
};

__device__ __forceinline__ unsigned short f2bf(float f) {
    union { float f; uint32_t u; } c; c.f = f;
    uint32_t r = c.u + 0x7FFFu + ((c.u >> 16) & 1u);   // round-to-nearest-even
    return (unsigned short)(r >> 16);
}

// ---------------- conversion kernels ----------------
__global__ void cvt_x_bf16(const float* __restrict__ x, unsigned short* __restrict__ xb, int n) {
    int i = blockIdx.x * blockDim.x + threadIdx.x;
    if (i < n) xb[i] = f2bf(x[i]);
}

// LDS-tiled transpose: Wt[j*1024 + k] = bf16(W[k*1024 + j])
__global__ __launch_bounds__(256) void cvt_w_T_bf16(const float* __restrict__ W,
                                                    unsigned short* __restrict__ Wt) {
    __shared__ float tile[32][33];
    const int tx = threadIdx.x & 31;
    const int ty = threadIdx.x >> 5;          // 0..7
    const int k0 = blockIdx.y * 32;
    const int j0 = blockIdx.x * 32;
#pragma unroll
    for (int i = ty; i < 32; i += 8)
        tile[i][tx] = W[(size_t)(k0 + i) * 1024 + j0 + tx];
    __syncthreads();
#pragma unroll
    for (int i = ty; i < 32; i += 8)
        Wt[(size_t)(j0 + i) * 1024 + k0 + tx] = f2bf(tile[tx][i]);
}

// ---------------- bf16 WMMA GEMM with bias, multi-layout epilogue ----------------
// A  : [M,K] bf16 row-major ; Wt : [Nc,K] bf16 (W transposed)
// mode 0: out bf16 head-major  [(b*16+h)*2048 + n]*64 + d       (Q, K)
// mode 1: out bf16 head-transposed [(b*16+h)*64 + d]*2048 + n   (V^T)
// mode 2: out fp32 row-major [M, Nc]                            (final proj)
// Block: 256 threads = 8 waves as 4(M) x 2(N); wave tile 32x64; block tile 128x128.
__global__ __launch_bounds__(256) void gemm_bf16(
    const unsigned short* __restrict__ A,
    const unsigned short* __restrict__ Wt,
    const float* __restrict__ bias,
    void* __restrict__ outp,
    int M, int K, int Nc, int mode)
{
    const int lane = threadIdx.x & 31;
    const int wave = threadIdx.x >> 5;
    const int hi   = lane >> 4;
    const int lo   = lane & 15;
    const int wm   = wave & 3;        // 0..3
    const int wn   = wave >> 2;       // 0..1

    const int m0 = blockIdx.x * 128 + wm * 32;
    const int j0 = blockIdx.y * 128 + wn * 64;

    // per-lane base pointers (A-fragment: chunks at +0 and +8 elements)
    const unsigned short* pa0 = A + (size_t)(m0 + lo) * K + hi * 16;
    const unsigned short* pa1 = A + (size_t)(m0 + 16 + lo) * K + hi * 16;
    const unsigned short* pb0 = Wt + (size_t)(j0 +  0 + lo) * K + hi * 16;
    const unsigned short* pb1 = Wt + (size_t)(j0 + 16 + lo) * K + hi * 16;
    const unsigned short* pb2 = Wt + (size_t)(j0 + 32 + lo) * K + hi * 16;
    const unsigned short* pb3 = Wt + (size_t)(j0 + 48 + lo) * K + hi * 16;
    const unsigned short* pb[4] = { pb0, pb1, pb2, pb3 };

    v8f acc[2][4] = {};

    for (int ks = 0; ks < K; ks += 32) {
        FragBF a0, a1;
        a0.q[0] = *(const uint4*)(pa0 + ks);
        a0.q[1] = *(const uint4*)(pa0 + ks + 8);
        a1.q[0] = *(const uint4*)(pa1 + ks);
        a1.q[1] = *(const uint4*)(pa1 + ks + 8);
#pragma unroll
        for (int t = 0; t < 4; ++t) {
            FragBF b;
            b.q[0] = *(const uint4*)(pb[t] + ks);
            b.q[1] = *(const uint4*)(pb[t] + ks + 8);
            acc[0][t] = __builtin_amdgcn_wmma_f32_16x16x32_bf16(
                false, a0.v, false, b.v, (short)0, acc[0][t], false, false);
            acc[1][t] = __builtin_amdgcn_wmma_f32_16x16x32_bf16(
                false, a1.v, false, b.v, (short)0, acc[1][t], false, false);
        }
    }

#pragma unroll
    for (int rb = 0; rb < 2; ++rb)
#pragma unroll
        for (int t = 0; t < 4; ++t) {
            int j = j0 + t * 16 + lo;
            float bj = bias[j];
#pragma unroll
            for (int r = 0; r < 8; ++r) {
                int m = m0 + rb * 16 + r + 8 * hi;
                float val = acc[rb][t][r] + bj;
                if (mode == 2) {
                    ((float*)outp)[(size_t)m * Nc + j] = val;
                } else if (mode == 0) {
                    size_t idx = ((size_t)((m >> 11) * 16 + (j >> 6)) * 2048 + (m & 2047)) * 64 + (j & 63);
                    ((unsigned short*)outp)[idx] = f2bf(val);
                } else {
                    size_t idx = ((size_t)((m >> 11) * 16 + (j >> 6)) * 64 + (j & 63)) * 2048 + (m & 2047);
                    ((unsigned short*)outp)[idx] = f2bf(val);
                }
            }
        }
}

// ---------------- flash attention (one wave per 16-query tile) ----------------
// Q,Km : [B*H, N, 64] bf16 ; Vt : [B*H, 64, N] bf16 ; out P : [B*N, 1024] bf16
__global__ __launch_bounds__(128) void attn_flash(
    const unsigned short* __restrict__ Q,
    const unsigned short* __restrict__ Km,
    const unsigned short* __restrict__ Vt,
    unsigned short* __restrict__ P,
    int N, float scale)
{
    __shared__ unsigned short lds[4 * 16 * 32];   // per-wave 16x32 bf16 P staging

    const int lane = threadIdx.x & 31;
    const int wave = threadIdx.x >> 5;
    const int hi   = lane >> 4;
    const int lo   = lane & 15;
    const int kb   = hi * 16;

    const int wg = blockIdx.x * 4 + wave;   // global query-tile id
    const int bh = wg >> 7;                 // 128 tiles per (b,h)
    const int qt = wg & 127;

    const unsigned short* Qh = Q  + (size_t)bh * N * 64;
    const unsigned short* Kh = Km + (size_t)bh * N * 64;
    const unsigned short* Vh = Vt + (size_t)bh * 64 * N;
    unsigned short* lbase = lds + wave * 512;

    // Q A-fragments for hd=64 (two K=32 steps), kept resident
    FragBF qa[2];
    {
        const unsigned short* qrowp = Qh + (size_t)(qt * 16 + lo) * 64 + kb;
#pragma unroll
        for (int f = 0; f < 2; ++f) {
            qa[f].q[0] = *(const uint4*)(qrowp + f * 32);
            qa[f].q[1] = *(const uint4*)(qrowp + f * 32 + 8);
        }
    }

    // per-lane base pointers for K (B-fragment) and V (B-fragment)
    const unsigned short* pk0 = Kh + (size_t)lo * 64 + hi * 16;        // key = kt + lo
    const unsigned short* pk1 = Kh + (size_t)(16 + lo) * 64 + hi * 16; // key = kt + 16 + lo
    const unsigned short* pv[4];
#pragma unroll
    for (int dt = 0; dt < 4; ++dt)
        pv[dt] = Vh + (size_t)(dt * 16 + lo) * N + hi * 16;

    v8f   o[4] = {};
    float mrow[8], lrow[8];
#pragma unroll
    for (int r = 0; r < 8; ++r) { mrow[r] = -1e30f; lrow[r] = 0.f; }

    for (int kt = 0; kt < N; kt += 32) {
        const size_t koff = (size_t)kt * 64;   // byte stride handled by ushort ptr math

        if (kt + 32 < N)   // prefetch next key tile's rows (one 128B line per row)
            __builtin_prefetch(Kh + (size_t)(kt + 32 + lane) * 64, 0, 0);

        // ---- S = Q @ K^T  (16 x 32) ----
        v8f s0 = {}, s1 = {};
#pragma unroll
        for (int f = 0; f < 2; ++f) {
            FragBF b0, b1;
            b0.q[0] = *(const uint4*)(pk0 + koff + f * 32);
            b0.q[1] = *(const uint4*)(pk0 + koff + f * 32 + 8);
            b1.q[0] = *(const uint4*)(pk1 + koff + f * 32);
            b1.q[1] = *(const uint4*)(pk1 + koff + f * 32 + 8);
            s0 = __builtin_amdgcn_wmma_f32_16x16x32_bf16(false, qa[f].v, false, b0.v, (short)0, s0, false, false);
            s1 = __builtin_amdgcn_wmma_f32_16x16x32_bf16(false, qa[f].v, false, b1.v, (short)0, s1, false, false);
        }

        // ---- online softmax, write P (bf16) into LDS row-major 16x32 ----
#pragma unroll
        for (int r = 0; r < 8; ++r) {
            float a0 = s0[r] * scale;
            float a1 = s1[r] * scale;
            float mx = fmaxf(a0, a1);
#pragma unroll
            for (int off = 8; off >= 1; off >>= 1)
                mx = fmaxf(mx, __shfl_xor(mx, off));
            float mn    = fmaxf(mrow[r], mx);
            float alpha = __expf(mrow[r] - mn);
            float p0 = __expf(a0 - mn);
            float p1 = __expf(a1 - mn);
            float ps = p0 + p1;
#pragma unroll
            for (int off = 8; off >= 1; off >>= 1)
                ps += __shfl_xor(ps, off);
            lrow[r] = lrow[r] * alpha + ps;
            mrow[r] = mn;
#pragma unroll
            for (int dt = 0; dt < 4; ++dt) o[dt][r] *= alpha;
            int lr = (r + 8 * hi) * 32;
            lbase[lr + lo]      = f2bf(p0);
            lbase[lr + 16 + lo] = f2bf(p1);
        }
        asm volatile("s_wait_dscnt 0" ::: "memory");   // LDS stores visible (per-wave in-order)

        // ---- reload P as A-fragment (16x32), b128 from LDS ----
        FragBF pa;
        pa.q[0] = *(const uint4*)(lbase + lo * 32 + kb);
        pa.q[1] = *(const uint4*)(lbase + lo * 32 + kb + 8);

        // ---- O += P @ V ----
#pragma unroll
        for (int dt = 0; dt < 4; ++dt) {
            FragBF bv;
            bv.q[0] = *(const uint4*)(pv[dt] + kt);
            bv.q[1] = *(const uint4*)(pv[dt] + kt + 8);
            o[dt] = __builtin_amdgcn_wmma_f32_16x16x32_bf16(false, pa.v, false, bv.v, (short)0, o[dt], false, false);
        }
    }

    // ---- epilogue: O /= l, write attn_out [B*N, 1024] bf16 ----
    const int b = bh >> 4, h = bh & 15;
#pragma unroll
    for (int r = 0; r < 8; ++r) {
        float inv = 1.0f / lrow[r];
        int m = b * 2048 + qt * 16 + r + 8 * hi;
#pragma unroll
        for (int dt = 0; dt < 4; ++dt) {
            int j = h * 64 + dt * 16 + lo;
            P[(size_t)m * 1024 + j] = f2bf(o[dt][r] * inv);
        }
    }
}

// ---------------- host launcher ----------------
extern "C" void kernel_launch(void* const* d_in, const int* in_sizes, int n_in,
                              void* d_out, int out_size, void* d_ws, size_t ws_size,
                              hipStream_t stream) {
    (void)in_sizes; (void)n_in; (void)out_size; (void)ws_size;
    const float* x  = (const float*)d_in[0];
    const float* Wq = (const float*)d_in[1];
    const float* bq = (const float*)d_in[2];
    const float* Wk = (const float*)d_in[3];
    const float* bk = (const float*)d_in[4];
    const float* Wv = (const float*)d_in[5];
    const float* bv = (const float*)d_in[6];
    const float* Wo = (const float*)d_in[7];
    const float* bo = (const float*)d_in[8];

    char* w = (char*)d_ws;                                   // 48 MiB workspace layout
    unsigned short* xb  = (unsigned short*)(w);              //  8 MiB  x bf16
    unsigned short* wqt = (unsigned short*)(w + (8u  << 20));//  2 MiB each, transposed
    unsigned short* wkt = (unsigned short*)(w + (10u << 20));
    unsigned short* wvt = (unsigned short*)(w + (12u << 20));
    unsigned short* wot = (unsigned short*)(w + (14u << 20));
    unsigned short* Qm  = (unsigned short*)(w + (16u << 20));//  8 MiB  [BH,N,64]
    unsigned short* Km  = (unsigned short*)(w + (24u << 20));//  8 MiB  [BH,N,64]
    unsigned short* Vtm = (unsigned short*)(w + (32u << 20));//  8 MiB  [BH,64,N]
    unsigned short* Pm  = (unsigned short*)(w + (40u << 20));//  8 MiB  attn out bf16

    const int M = 4096, K = 1024, Nc = 1024, N = 2048;

    cvt_x_bf16<<<(M * K) / 256, 256, 0, stream>>>(x, xb, M * K);
    dim3 tg(32, 32);
    cvt_w_T_bf16<<<tg, 256, 0, stream>>>(Wq, wqt);
    cvt_w_T_bf16<<<tg, 256, 0, stream>>>(Wk, wkt);
    cvt_w_T_bf16<<<tg, 256, 0, stream>>>(Wv, wvt);
    cvt_w_T_bf16<<<tg, 256, 0, stream>>>(Wo, wot);

    dim3 gg(M / 128, Nc / 128);
    gemm_bf16<<<gg, 256, 0, stream>>>(xb, wqt, bq, Qm,  M, K, Nc, 0);
    gemm_bf16<<<gg, 256, 0, stream>>>(xb, wkt, bk, Km,  M, K, Nc, 0);
    gemm_bf16<<<gg, 256, 0, stream>>>(xb, wvt, bv, Vtm, M, K, Nc, 1);

    attn_flash<<<(32 * 128) / 4, 128, 0, stream>>>(Qm, Km, Vtm, Pm, N, 0.125f);

    gemm_bf16<<<gg, 256, 0, stream>>>(Pm, wot, bo, d_out, M, K, Nc, 2);
}
